// GAT_69973607186512
// MI455X (gfx1250) — matrix operand order repaired and checked by reading, not compiled, not run on hardware
//
#include <hip/hip_runtime.h>
#include <hip/hip_bf16.h>
#include <math.h>

// ---------------------------------------------------------------------------
// GAT 2-layer forward for MI455X (gfx1250, wave32, WMMA bf16)
// ---------------------------------------------------------------------------

#define N_NODES   100000
#define N_EDGES   1600000
#define IN_FEATS  256
#define HIDDEN    128
#define N_CLASSES 128
#define NEG_SLOPE 0.2f

typedef __attribute__((ext_vector_type(16))) __bf16 v16bf;
typedef __attribute__((ext_vector_type(8)))  __bf16 v8bf;
typedef __attribute__((ext_vector_type(8)))  float  v8f;

// ---------------------------------------------------------------------------
// GEMM: H[N,128] = X[N,KD] @ W[128,KD]^T   (bf16 WMMA, fp32 accumulate)
// block = 256 threads = 8 waves; block -> 16-row node tile; wave -> 16 cols.
// ---------------------------------------------------------------------------
template <int KD>
__global__ __launch_bounds__(256) void gemm_wmma_bf16(
    const float* __restrict__ X, const float* __restrict__ W,
    float* __restrict__ H) {
  __shared__ __bf16 sA[16][KD + 8];  // +8 bf16 pad -> conflict-free rows

  const int tid     = threadIdx.x;
  const int rowbase = blockIdx.x * 16;

  // Cooperative load of the 16xKD A tile, fp32 -> bf16, coalesced over cols.
  for (int i = tid; i < 16 * KD; i += 256) {
    const int r = i / KD;
    const int c = i - r * KD;
    sA[r][c] = (__bf16)X[(size_t)(rowbase + r) * KD + c];
  }
  __syncthreads();

  const int wave = tid >> 5;
  const int lane = tid & 31;
  const int m    = lane & 15;   // A row within tile / B column within tile
  const int half = lane >> 4;
  const int wrow = wave * 16 + m;  // output-feature index (row of W)

  v8f acc = {};

#pragma unroll
  for (int k0 = 0; k0 < KD; k0 += 32) {
    // ---- A fragment (16x32 bf16): two aligned 16B runs from LDS ----
    const v8bf a0 = *(const v8bf*)&sA[m][k0 + half * 16];
    const v8bf a1 = *(const v8bf*)&sA[m][k0 + half * 16 + 8];
    v16bf a;
#pragma unroll
    for (int j = 0; j < 8; ++j) { a[j] = a0[j]; a[8 + j] = a1[j]; }

    // ---- B fragment (32x16 bf16): 16 contiguous fp32 from W (L2-hot) ----
    const float4* pb = (const float4*)&W[(size_t)wrow * KD + k0 + half * 16];
    v16bf b;
#pragma unroll
    for (int q = 0; q < 4; ++q) {
      const float4 t = pb[q];
      b[4 * q + 0] = (__bf16)t.x;
      b[4 * q + 1] = (__bf16)t.y;
      b[4 * q + 2] = (__bf16)t.z;
      b[4 * q + 3] = (__bf16)t.w;
    }

    acc = __builtin_amdgcn_wmma_f32_16x16x32_bf16(
        /*neg_a=*/false, a, /*neg_b=*/false, b,
        /*c_mod=*/(short)0, acc, /*reuse_a=*/false, /*reuse_b=*/false);
  }

  // C/D layout: lane column m, VGPR v -> row (v + 8*half)
#pragma unroll
  for (int v = 0; v < 8; ++v) {
    H[(size_t)(rowbase + v + half * 8) * 128 + wave * 16 + m] = acc[v];
  }
}

// ---------------------------------------------------------------------------
// Per-node attention scores: el[n] = h[n].al, er[n] = h[n].ar  (D = 128)
// one wave per node, 8 nodes per block
// ---------------------------------------------------------------------------
__global__ __launch_bounds__(256) void node_scores(
    const float* __restrict__ H, const float* __restrict__ al,
    const float* __restrict__ ar, float* __restrict__ el,
    float* __restrict__ er) {
  const int wave = threadIdx.x >> 5;
  const int lane = threadIdx.x & 31;
  const int n    = blockIdx.x * 8 + wave;
  const float* h = H + (size_t)n * 128;

  float sl = 0.f, sr = 0.f;
#pragma unroll
  for (int j = 0; j < 4; ++j) {
    const int   k = lane + 32 * j;
    const float v = h[k];
    sl += v * al[k];
    sr += v * ar[k];
  }
#pragma unroll
  for (int off = 16; off > 0; off >>= 1) {
    sl += __shfl_xor(sl, off, 32);
    sr += __shfl_xor(sr, off, 32);
  }
  if (lane == 0) { el[n] = sl; er[n] = sr; }
}

// ---------------------------------------------------------------------------
// init: out[n,f] = bias[f];  m[n] = -inf;  s[n] = 0     (grid over N*128)
// ---------------------------------------------------------------------------
__global__ __launch_bounds__(256) void init_out(
    const float* __restrict__ bias, float* __restrict__ out,
    float* __restrict__ m, float* __restrict__ s) {
  const int i   = blockIdx.x * 256 + threadIdx.x;
  const int col = i & 127;
  out[i] = bias[col];
  if (col == 0) {
    const int n = i >> 7;
    m[n] = -__builtin_inff();
    s[n] = 0.f;
  }
}

// float atomic-max via signed-max / unsigned-min ordering trick
__device__ __forceinline__ void atomicMaxF(float* addr, float val) {
  if (val >= 0.f)
    atomicMax((int*)addr, __float_as_int(val));
  else
    atomicMin((unsigned int*)addr, (unsigned int)__float_as_int(val));
}

// ---------------------------------------------------------------------------
// edge pass 1: e = leaky_relu(el[src]+er[dst]); m[dst] = max(m[dst], e)
// ---------------------------------------------------------------------------
__global__ __launch_bounds__(256) void edge_max(
    const int* __restrict__ src, const int* __restrict__ dst,
    const float* __restrict__ el, const float* __restrict__ er,
    float* __restrict__ ebuf, float* __restrict__ m) {
  const int e  = blockIdx.x * 256 + threadIdx.x;
  const int s_ = src[e];
  const int d_ = dst[e];
  float v = el[s_] + er[d_];
  v = v > 0.f ? v : NEG_SLOPE * v;
  ebuf[e] = v;
  atomicMaxF(&m[d_], v);
}

// ---------------------------------------------------------------------------
// edge pass 2: ex = exp(e - m[dst]); s[dst] += ex
// ---------------------------------------------------------------------------
__global__ __launch_bounds__(256) void edge_expsum(
    const int* __restrict__ dst, const float* __restrict__ ebuf,
    const float* __restrict__ m, float* __restrict__ exb,
    float* __restrict__ s) {
  const int e  = blockIdx.x * 256 + threadIdx.x;
  const int d_ = dst[e];
  const float ex = __expf(ebuf[e] - m[d_]);
  exb[e] = ex;
  atomicAdd(&s[d_], ex);
}

// ---------------------------------------------------------------------------
// edge pass 3: out[dst,:] += (ex/s[dst]) * h[src,:]   (128 lanes per edge,
// 2 edges per block -> coalesced gather + coalesced fp32 atomics)
// ---------------------------------------------------------------------------
__global__ __launch_bounds__(256) void edge_scatter(
    const int* __restrict__ src, const int* __restrict__ dst,
    const float* __restrict__ exb, const float* __restrict__ s,
    const float* __restrict__ H, float* __restrict__ out) {
  const int e  = blockIdx.x * 2 + (threadIdx.x >> 7);
  const int f  = threadIdx.x & 127;
  const int s_ = src[e];
  const int d_ = dst[e];
  const float alpha = exb[e] / s[d_];
  atomicAdd(&out[(size_t)d_ * 128 + f], alpha * H[(size_t)s_ * 128 + f]);
}

// ---------------------------------------------------------------------------
// elementwise ELU in place (alpha = 1)
// ---------------------------------------------------------------------------
__global__ __launch_bounds__(256) void elu_ip(float* __restrict__ x) {
  const int i = blockIdx.x * 256 + threadIdx.x;
  const float v = x[i];
  x[i] = v > 0.f ? v : (__expf(v) - 1.f);
}

// ---------------------------------------------------------------------------
// host orchestration
// ---------------------------------------------------------------------------
extern "C" void kernel_launch(void* const* d_in, const int* in_sizes, int n_in,
                              void* d_out, int out_size, void* d_ws,
                              size_t ws_size, hipStream_t stream) {
  (void)in_sizes; (void)n_in; (void)out_size; (void)ws_size;

  const float* feats = (const float*)d_in[0];
  const int*   src   = (const int*)d_in[1];
  const int*   dst   = (const int*)d_in[2];
  const float* W1    = (const float*)d_in[3];
  const float* al1   = (const float*)d_in[4];
  const float* ar1   = (const float*)d_in[5];
  const float* b1    = (const float*)d_in[6];
  const float* W2    = (const float*)d_in[7];
  const float* al2   = (const float*)d_in[8];
  const float* ar2   = (const float*)d_in[9];
  const float* b2    = (const float*)d_in[10];
  float*       out   = (float*)d_out;

  // workspace layout (floats)
  float* ws   = (float*)d_ws;
  float* h    = ws;                                   // N*128
  float* act  = h + (size_t)N_NODES * 128;            // N*128
  float* el   = act + (size_t)N_NODES * 128;          // N
  float* er   = el + N_NODES;                         // N
  float* m    = er + N_NODES;                         // N
  float* s    = m + N_NODES;                          // N
  float* ebuf = s + N_NODES;                          // E
  float* exb  = ebuf + N_EDGES;                       // E

  const dim3 blk(256);
  const int gGemm    = N_NODES / 16;        // 6250
  const int gNode    = N_NODES / 8;         // 12500
  const int gNF      = N_NODES * 128 / 256; // 50000
  const int gEdge    = N_EDGES / 256;       // 6250
  const int gScatter = N_EDGES / 2;         // 800000

  // ---------------- layer 1 ----------------
  gemm_wmma_bf16<IN_FEATS><<<gGemm, blk, 0, stream>>>(feats, W1, h);
  node_scores<<<gNode, blk, 0, stream>>>(h, al1, ar1, el, er);
  init_out<<<gNF, blk, 0, stream>>>(b1, act, m, s);
  edge_max<<<gEdge, blk, 0, stream>>>(src, dst, el, er, ebuf, m);
  edge_expsum<<<gEdge, blk, 0, stream>>>(dst, ebuf, m, exb, s);
  edge_scatter<<<gScatter, blk, 0, stream>>>(src, dst, exb, s, h, act);
  elu_ip<<<gNF, blk, 0, stream>>>(act);

  // ---------------- layer 2 ----------------
  gemm_wmma_bf16<HIDDEN><<<gGemm, blk, 0, stream>>>(act, W2, h);
  node_scores<<<gNode, blk, 0, stream>>>(h, al2, ar2, el, er);
  init_out<<<gNF, blk, 0, stream>>>(b2, out, m, s);
  edge_max<<<gEdge, blk, 0, stream>>>(src, dst, el, er, ebuf, m);
  edge_expsum<<<gEdge, blk, 0, stream>>>(dst, ebuf, m, exb, s);
  edge_scatter<<<gScatter, blk, 0, stream>>>(src, dst, exb, s, h, out);
}
